// DynamicQuantizedLinear_89816356093955
// MI455X (gfx1250) — matrix operand order repaired and checked by reading, not compiled, not run on hardware
//
#include <hip/hip_runtime.h>
#include <stdint.h>

typedef _Float16 f16;
typedef __attribute__((ext_vector_type(8)))  _Float16 v8h;
typedef __attribute__((ext_vector_type(16))) _Float16 v16h;
typedef __attribute__((ext_vector_type(8)))  float    v8f;
typedef __attribute__((ext_vector_type(8)))  int8_t   c8;

// Problem sizes (from reference: B=4, S=512, IN_F=4096, OUT_F=11008)
static constexpr int M_DIM = 2048;   // B*S
static constexpr int K_DIM = 4096;   // IN_F
static constexpr int N_DIM = 11008;  // OUT_F

// Tiling: 128x256 block tile, 8 waves of 64x64, K staged 64 at a time.
static constexpr int BM = 128;
static constexpr int BN = 256;
static constexpr int BK = 64;
static constexpr int LDA = BK + 8;   // padded LDS row stride (halves): 144B, conflict-free, 16B aligned

__global__ __launch_bounds__(256)
void qlinear_wmma_kernel(const f16* __restrict__ X,      // [M,K] fp16
                         const int8_t* __restrict__ Wq,  // [N,K] int8
                         const f16* __restrict__ Sc,     // [N] fp16 scale
                         const f16* __restrict__ Bi,     // [N] fp16 bias
                         f16* __restrict__ Out)          // [M,N] fp16
{
    __shared__ __align__(16) f16 As[BM][LDA];  // x tile               (18.4 KB)
    __shared__ __align__(16) f16 Bs[BN][LDA];  // weights as fp16 ints (36.9 KB)

    const int tid  = threadIdx.x;
    const int lane = tid & 31;
    const int wave = tid >> 5;       // 0..7
    const int wm   = wave >> 2;      // 0..1 : 64-row slab
    const int wn   = wave & 3;       // 0..3 : 64-col slab
    const int lr   = lane & 15;
    const int g    = lane >> 4;

    const int mBlock = blockIdx.y * BM;
    const int nBlock = blockIdx.x * BN;

    v8f acc[4][4];
    const v8f vzero = {};
#pragma unroll
    for (int mt = 0; mt < 4; ++mt)
#pragma unroll
        for (int nt = 0; nt < 4; ++nt)
            acc[mt][nt] = vzero;

    for (int k0 = 0; k0 < K_DIM; k0 += BK) {
        // ---- stage A: 128 rows x 64 halves, 4 x b128 per thread ----
#pragma unroll
        for (int i = 0; i < 4; ++i) {
            const int v = tid + i * 256;       // 0..1023
            const int r = v >> 3;              // 0..127
            const int c = (v & 7) << 3;        // half offset 0..56
            *(uint4*)&As[r][c] =
                *(const uint4*)(X + (size_t)(mBlock + r) * K_DIM + k0 + c);
        }
        // ---- stage B: 256 rows x 64 int8 -> fp16, 4 x b128 per thread ----
#pragma unroll
        for (int i = 0; i < 4; ++i) {
            const int v = tid + i * 256;       // 0..1023
            const int r = v >> 2;              // 0..255
            const int c = (v & 3) << 4;        // elem offset 0..48
            union { uint4 u; c8 b[2]; } q;
            q.u = *(const uint4*)(Wq + (size_t)(nBlock + r) * K_DIM + k0 + c);
            *(v8h*)&Bs[r][c]     = __builtin_convertvector(q.b[0], v8h);
            *(v8h*)&Bs[r][c + 8] = __builtin_convertvector(q.b[1], v8h);
        }
        // ---- prefetch next K chunk (global_prefetch_b8) ----
        if (k0 + BK < K_DIM) {
            __builtin_prefetch(X  + (size_t)(mBlock + (tid >> 1)) * K_DIM + k0 + BK + (tid & 1) * 32, 0, 0);
            __builtin_prefetch(Wq + (size_t)(nBlock + tid) * K_DIM + k0 + BK, 0, 0);
        }
        __syncthreads();

        // ---- compute: 2 k-steps of 32; 16 WMMAs per k-step (1 ds_load_b128 per WMMA) ----
#pragma unroll
        for (int kk = 0; kk < BK; kk += 32) {
            v16h a[4], b[4];
#pragma unroll
            for (int mt = 0; mt < 4; ++mt) {
                const f16* ar = &As[wm * 64 + mt * 16 + lr][0];
                v8h lo = *(const v8h*)(ar + kk + g * 8);        // e0..7 : K = g*8 + e
                v8h hi = *(const v8h*)(ar + kk + 16 + g * 8);   // e8..15: K = 16 + g*8 + (e-8)
                a[mt] = __builtin_shufflevector(lo, hi,
                        0,1,2,3,4,5,6,7,8,9,10,11,12,13,14,15);
            }
#pragma unroll
            for (int nt = 0; nt < 4; ++nt) {
                const f16* br = &Bs[wn * 64 + nt * 16 + lr][0];
                v8h lo = *(const v8h*)(br + kk + g * 8);        // e0..7 : K = g*8 + e
                v8h hi = *(const v8h*)(br + kk + 16 + g * 8);   // e8..15: K = 16 + g*8 + (e-8)
                b[nt] = __builtin_shufflevector(lo, hi,
                        0,1,2,3,4,5,6,7,8,9,10,11,12,13,14,15);
            }
#pragma unroll
            for (int mt = 0; mt < 4; ++mt)
#pragma unroll
                for (int nt = 0; nt < 4; ++nt)
                    acc[mt][nt] = __builtin_amdgcn_wmma_f32_16x16x32_f16(
                        false, a[mt], false, b[nt],
                        (short)0, acc[mt][nt], false, false);
        }
        __syncthreads();
    }

    // ---- epilogue: out = acc * scale[n] + bias[n] ----
#pragma unroll
    for (int nt = 0; nt < 4; ++nt) {
        const int n = nBlock + wn * 64 + nt * 16 + lr;
        const float sc = (float)Sc[n];
        const float bs = (float)Bi[n];
#pragma unroll
        for (int mt = 0; mt < 4; ++mt) {
            const int mBase = mBlock + wm * 64 + mt * 16 + g * 8;
#pragma unroll
            for (int r = 0; r < 8; ++r) {
                Out[(size_t)(mBase + r) * N_DIM + n] =
                    (f16)(acc[mt][nt][r] * sc + bs);
            }
        }
    }
}

extern "C" void kernel_launch(void* const* d_in, const int* in_sizes, int n_in,
                              void* d_out, int out_size, void* d_ws, size_t ws_size,
                              hipStream_t stream) {
    const f16*    x  = (const f16*)d_in[0];
    const int8_t* wq = (const int8_t*)d_in[1];
    const f16*    sc = (const f16*)d_in[2];
    const f16*    bi = (const f16*)d_in[3];
    f16*          out = (f16*)d_out;

    dim3 grid(N_DIM / BN, M_DIM / BM);  // 43 x 16
    dim3 block(256);
    qlinear_wmma_kernel<<<grid, block, 0, stream>>>(x, wq, sc, bi, out);

    (void)in_sizes; (void)n_in; (void)out_size; (void)d_ws; (void)ws_size;
}
